// SlotAttention_62096637165994
// MI455X (gfx1250) — matrix operand order repaired and checked by reading, not compiled, not run on hardware
//
#include <hip/hip_runtime.h>
#include <hip/hip_bf16.h>

typedef __bf16 v16bf __attribute__((ext_vector_type(16)));
typedef __bf16 v8bf  __attribute__((ext_vector_type(8)));
typedef float  v8f   __attribute__((ext_vector_type(8)));

union AFrag { v16bf v; v8bf h[2]; __bf16 e[16]; };

#define WMMA_BF16(A, B, C) \
  __builtin_amdgcn_wmma_f32_16x16x32_bf16(false, (A), false, (B), (short)0, (C), false, false)

constexpr int   B_ = 32, N_ = 16384, D_ = 64, K_ = 8;
constexpr float LN_EPS_ = 1e-3f, ATTN_EPS_ = 1e-8f;

// ---------------------------------------------------------------------------
// slots = loc + exp(log_scl)*noise ; zero sumval / raw / colsum accumulators
// ---------------------------------------------------------------------------
__global__ void init_kernel(const float* __restrict__ noise, const float* __restrict__ loc,
                            const float* __restrict__ lscl, float* __restrict__ slots,
                            float* __restrict__ sumval, float* __restrict__ raw,
                            float* __restrict__ colsum) {
  int idx = blockIdx.x * 256 + threadIdx.x;   // 16384 threads
  if (idx < B_ * K_ * D_) {
    int d = idx & 63;
    slots[idx] = loc[d] + __expf(lscl[d]) * noise[idx];
    raw[idx] = 0.f;
  }
  if (idx < B_ * D_) sumval[idx] = 0.f;
  if (idx < B_ * K_) colsum[idx] = 0.f;
}

// ---------------------------------------------------------------------------
// x_n = LN(x); x_key = x_n@Wk (row-major bf16); x_valT = (x_n@Wv)^T (bf16);
// sumval[b][d] += sum_n x_val[n][d]
// grid 4096 x 256 : block = 128 token rows (one batch each), wave = 16 rows
// ---------------------------------------------------------------------------
__global__ __launch_bounds__(256) void proj_kernel(
    const float* __restrict__ x, const float* __restrict__ g_in, const float* __restrict__ b_in,
    const float* __restrict__ Wk, const float* __restrict__ Wv,
    __bf16* __restrict__ xkey, __bf16* __restrict__ xvalT, float* __restrict__ sumval) {
  __shared__ __bf16 wkT[64 * 64];           // [col][row]  (B-frag friendly)
  __shared__ __bf16 wvT[64 * 64];
  __shared__ __bf16 tile[8][16 * 64];       // per-wave LN'd bf16 tile
  __shared__ float  lng[64], lnb[64], svsh[64];

  int tid = threadIdx.x, wave = tid >> 5, lane = tid & 31;
  for (int idx = tid; idx < 4096; idx += 256) {
    int i = idx >> 6, j = idx & 63;
    wkT[j * 64 + i] = (__bf16)Wk[idx];
    wvT[j * 64 + i] = (__bf16)Wv[idx];
  }
  if (tid < 64) { lng[tid] = g_in[tid]; lnb[tid] = b_in[tid]; svsh[tid] = 0.f; }
  __syncthreads();

  // LayerNorm: 2 lanes per row (32 elems each)
  int r = lane >> 1, half = lane & 1;
  size_t row = (size_t)blockIdx.x * 128 + wave * 16 + r;
  const float* xr = x + row * 64 + half * 32;
  float v[32], s = 0.f, sq = 0.f;
  for (int i = 0; i < 32; ++i) { float t = xr[i]; v[i] = t; s += t; sq += t * t; }
  s += __shfl_xor(s, 1);  sq += __shfl_xor(sq, 1);
  float mu = s * (1.f / 64.f);
  float rstd = rsqrtf(sq * (1.f / 64.f) - mu * mu + LN_EPS_);
  __bf16* trow = &tile[wave][r * 64 + half * 32];
  for (int i = 0; i < 32; ++i) {
    int c = half * 32 + i;
    trow[i] = (__bf16)((v[i] - mu) * rstd * lng[c] + lnb[c]);
  }
  __syncthreads();

  int hs = lane >> 4, rr = lane & 15;
  AFrag a0, a1;                              // A 16x32 frags, K=0..31 / 32..63
  {
    const __bf16* base = &tile[wave][rr * 64];
    a0.h[0] = *(const v8bf*)(base + hs * 16);
    a0.h[1] = *(const v8bf*)(base + hs * 16 + 8);
    a1.h[0] = *(const v8bf*)(base + 32 + hs * 16);
    a1.h[1] = *(const v8bf*)(base + 32 + hs * 16 + 8);
  }
  int b  = blockIdx.x >> 7;                            // batch (128 blocks/batch)
  int n0 = ((blockIdx.x & 127) * 128) + wave * 16;     // token base in batch
  for (int ct = 0; ct < 4; ++ct) {
    int col = ct * 16 + rr;
    AFrag bk0, bk1, bv0, bv1;
    const __bf16* kb = &wkT[col * 64];
    bk0.h[0] = *(const v8bf*)(kb + hs * 16);       bk0.h[1] = *(const v8bf*)(kb + hs * 16 + 8);
    bk1.h[0] = *(const v8bf*)(kb + 32 + hs * 16);  bk1.h[1] = *(const v8bf*)(kb + 32 + hs * 16 + 8);
    const __bf16* vb = &wvT[col * 64];
    bv0.h[0] = *(const v8bf*)(vb + hs * 16);       bv0.h[1] = *(const v8bf*)(vb + hs * 16 + 8);
    bv1.h[0] = *(const v8bf*)(vb + 32 + hs * 16);  bv1.h[1] = *(const v8bf*)(vb + 32 + hs * 16 + 8);
    v8f ak = {}; ak = WMMA_BF16(a0.v, bk0.v, ak); ak = WMMA_BF16(a1.v, bk1.v, ak);
    v8f av = {}; av = WMMA_BF16(a0.v, bv0.v, av); av = WMMA_BF16(a1.v, bv1.v, av);
    float sv = 0.f;
    for (int j = 0; j < 8; ++j) {                 // D: row = j + hs*8, col = col
      int tokl = j + hs * 8;
      size_t grow = (size_t)b * N_ + n0 + tokl;
      xkey[grow * 64 + col] = (__bf16)ak[j];
      xvalT[((size_t)b * 64 + col) * N_ + (n0 + tokl)] = (__bf16)av[j];
      sv += av[j];
    }
    atomicAdd(&svsh[col], sv);
  }
  __syncthreads();
  if (tid < 64) atomicAdd(&sumval[b * 64 + tid], svsh[tid]);
}

// ---------------------------------------------------------------------------
// Fused per-step pass: each block computes its batch's q tile (LN + matvec)
// into LDS, then streams 1024 tokens: logits via WMMA, softmax over 8 slots
// (shfl_xor), updates via WMMA (softmax D-frag feeds updates A-frag directly).
// grid = B*16 blocks x 256
// ---------------------------------------------------------------------------
__global__ __launch_bounds__(256) void attn_kernel(
    const float* __restrict__ slots, const float* __restrict__ gq, const float* __restrict__ bq,
    const float* __restrict__ Wq,
    const __bf16* __restrict__ xkey, const __bf16* __restrict__ xvalT,
    float* __restrict__ raw, float* __restrict__ colsum) {
  __shared__ float  rawsh[8 * 64];
  __shared__ float  ssh[8][64];     // LN'd slot rows
  __shared__ __bf16 qsh[16][64];    // q, zero-padded to 16 slot rows
  int tid = threadIdx.x, wave = tid >> 5, lane = tid & 31;
  int hs = lane >> 4, cl = lane & 15;
  int b = blockIdx.x >> 4, pib = blockIdx.x & 15;
  for (int i = tid; i < 512; i += 256) rawsh[i] = 0.f;

  // ---- q projection (redundant per block, trivial): wave w handles slot w
  {
    const float* srow = slots + ((size_t)b * 8 + wave) * 64;
    float s0 = srow[lane], s1 = srow[lane + 32];
    float s = s0 + s1, sq = s0 * s0 + s1 * s1;
    s  += __shfl_xor(s, 1);  s  += __shfl_xor(s, 2);  s  += __shfl_xor(s, 4);
    s  += __shfl_xor(s, 8);  s  += __shfl_xor(s, 16);
    sq += __shfl_xor(sq, 1); sq += __shfl_xor(sq, 2); sq += __shfl_xor(sq, 4);
    sq += __shfl_xor(sq, 8); sq += __shfl_xor(sq, 16);
    float mu = s * (1.f / 64.f);
    float rstd = rsqrtf(sq * (1.f / 64.f) - mu * mu + LN_EPS_);
    ssh[wave][lane]      = (s0 - mu) * rstd * gq[lane]      + bq[lane];
    ssh[wave][lane + 32] = (s1 - mu) * rstd * gq[lane + 32] + bq[lane + 32];
    qsh[wave + 8][lane] = (__bf16)0.0f;          // zero pad rows 8..15
    qsh[wave + 8][lane + 32] = (__bf16)0.0f;
  }
  __syncthreads();
  {
    float a0 = 0.f, a1 = 0.f;
    for (int i = 0; i < 64; ++i) {
      float si = ssh[wave][i];
      a0 += si * Wq[i * 64 + lane];
      a1 += si * Wq[i * 64 + lane + 32];
    }
    qsh[wave][lane]      = (__bf16)(a0 * 0.125f);   // * 1/sqrt(64)
    qsh[wave][lane + 32] = (__bf16)(a1 * 0.125f);
  }
  __syncthreads();

  AFrag q0, q1;                                    // B-frags of q^T (64 x 16)
  {
    const __bf16* qb = &qsh[cl][0];
    q0.h[0] = *(const v8bf*)(qb + hs * 16);      q0.h[1] = *(const v8bf*)(qb + hs * 16 + 8);
    q1.h[0] = *(const v8bf*)(qb + 32 + hs * 16); q1.h[1] = *(const v8bf*)(qb + 32 + hs * 16 + 8);
  }
  v8f upd[4] = {{}, {}, {}, {}};
  float csum = 0.f;
  bool validc = (cl < 8);
  int tokbase = pib * 1024 + wave * 128;
  for (int it = 0; it < 4; ++it) {
    int t0 = tokbase + it * 32;
    v8f pt[2];
    for (int tile = 0; tile < 2; ++tile) {
      int n = t0 + tile * 16 + cl;                 // token row (A-frag row)
      const __bf16* kb = xkey + ((size_t)b * N_ + n) * 64;
      AFrag a0, a1;
      a0.h[0] = *(const v8bf*)(kb + hs * 16);      a0.h[1] = *(const v8bf*)(kb + hs * 16 + 8);
      a1.h[0] = *(const v8bf*)(kb + 32 + hs * 16); a1.h[1] = *(const v8bf*)(kb + 32 + hs * 16 + 8);
      v8f lg = {};
      lg = WMMA_BF16(a0.v, q0.v, lg);
      lg = WMMA_BF16(a1.v, q1.v, lg);
      v8f p;
      for (int j = 0; j < 8; ++j) {                // softmax over slot columns
        float vv = validc ? lg[j] : -1e30f;
        float mx = vv;
        mx = fmaxf(mx, __shfl_xor(mx, 1));
        mx = fmaxf(mx, __shfl_xor(mx, 2));
        mx = fmaxf(mx, __shfl_xor(mx, 4));
        float e = validc ? __expf(lg[j] - mx) : 0.f;
        float sm = e;
        sm += __shfl_xor(sm, 1);
        sm += __shfl_xor(sm, 2);
        sm += __shfl_xor(sm, 4);
        float pv = validc ? e / sm : 0.f;
        p[j] = pv;
        csum += pv;
      }
      pt[tile] = p;
    }
    AFrag ap;                                      // A-frag of P^T (16 x 32 tokens)
    for (int j = 0; j < 8; ++j) { ap.e[j] = (__bf16)pt[0][j]; ap.e[8 + j] = (__bf16)pt[1][j]; }
    for (int dt = 0; dt < 4; ++dt) {               // P^T @ x_val -> slots x d
      int d = dt * 16 + cl;
      const __bf16* vb = xvalT + ((size_t)b * 64 + d) * N_ + t0 + hs * 8;
      AFrag bv;
      bv.h[0] = *(const v8bf*)vb;
      bv.h[1] = *(const v8bf*)(vb + 16);
      upd[dt] = WMMA_BF16(ap.v, bv.v, upd[dt]);
    }
  }
  csum += __shfl_xor(csum, 16);
  if (lane < 8) atomicAdd(&colsum[b * 8 + lane], csum);
  for (int dt = 0; dt < 4; ++dt)
    for (int j = 0; j < 8; ++j) {
      int slot = j + hs * 8;
      if (slot < 8) atomicAdd(&rawsh[slot * 64 + dt * 16 + cl], upd[dt][j]);
    }
  __syncthreads();
  for (int i = tid; i < 512; i += 256) atomicAdd(&raw[(size_t)b * 512 + i], rawsh[i]);
}

// ---------------------------------------------------------------------------
// updates -> GRU -> LN -> MLP residual; re-zeroes its accumulators for the
// next step.  grid = B*8 blocks x 64 threads
// ---------------------------------------------------------------------------
__global__ __launch_bounds__(64) void combine_kernel(
    float* __restrict__ raw, float* __restrict__ colsum, const float* __restrict__ sumval,
    float* __restrict__ slots,
    const float* __restrict__ Wg, const float* __restrict__ Ug,
    const float* __restrict__ bgin, const float* __restrict__ bgrec,
    const float* __restrict__ glin, const float* __restrict__ blin,
    const float* __restrict__ W1, const float* __restrict__ b1,
    const float* __restrict__ W2, const float* __restrict__ b2) {
  int blk = blockIdx.x, b = blk >> 3, t = threadIdx.x;
  __shared__ float ush[64], spsh[64], snsh[64], msh[64], hsh[128];
  float cs = colsum[blk] + ATTN_EPS_ * (float)N_;
  float u  = (raw[(size_t)blk * 64 + t] + ATTN_EPS_ * sumval[b * 64 + t]) / cs;
  raw[(size_t)blk * 64 + t] = 0.f;          // ready for next step
  if (t == 0) colsum[blk] = 0.f;
  float sp = slots[(size_t)blk * 64 + t];
  ush[t] = u;  spsh[t] = sp;
  __syncthreads();
  float gz = bgin[t],       hz = bgrec[t];
  float gr = bgin[64 + t],  hr = bgrec[64 + t];
  float gh = bgin[128 + t], hh = bgrec[128 + t];
  for (int i = 0; i < 64; ++i) {
    float ui = ush[i], si = spsh[i];
    const float* wr = Wg + i * 192;
    const float* ur = Ug + i * 192;
    gz += ui * wr[t]; gr += ui * wr[64 + t]; gh += ui * wr[128 + t];
    hz += si * ur[t]; hr += si * ur[64 + t]; hh += si * ur[128 + t];
  }
  float z = 1.f / (1.f + __expf(-(gz + hz)));
  float r = 1.f / (1.f + __expf(-(gr + hr)));
  float sn = z * sp + (1.f - z) * tanhf(gh + r * hh);
  snsh[t] = sn;
  __syncthreads();
  float s = 0.f, sq = 0.f;
  for (int i = 0; i < 64; ++i) { float w = snsh[i]; s += w; sq += w * w; }
  float mu = s / 64.f;
  msh[t] = (sn - mu) * rsqrtf(sq / 64.f - mu * mu + LN_EPS_) * glin[t] + blin[t];
  __syncthreads();
  for (int h = t; h < 128; h += 64) {
    float a = b1[h];
    for (int i = 0; i < 64; ++i) a += msh[i] * W1[i * 128 + h];
    hsh[h] = fmaxf(a, 0.f);
  }
  __syncthreads();
  float o = b2[t];
  for (int h2 = 0; h2 < 128; ++h2) o += hsh[h2] * W2[h2 * 64 + t];
  slots[(size_t)blk * 64 + t] = sn + o;
}

// ---------------------------------------------------------------------------
extern "C" void kernel_launch(void* const* d_in, const int* in_sizes, int n_in,
                              void* d_out, int out_size, void* d_ws, size_t ws_size,
                              hipStream_t stream) {
  const float* x        = (const float*)d_in[0];
  const float* noise    = (const float*)d_in[1];
  const float* ln_in_g  = (const float*)d_in[2];
  const float* ln_in_b  = (const float*)d_in[3];
  const float* W_key    = (const float*)d_in[4];
  const float* W_val    = (const float*)d_in[5];
  const float* ln_q_g   = (const float*)d_in[6];
  const float* ln_q_b   = (const float*)d_in[7];
  const float* W_qry    = (const float*)d_in[8];
  const float* loc      = (const float*)d_in[9];
  const float* log_scl  = (const float*)d_in[10];
  const float* W_gru    = (const float*)d_in[11];
  const float* U_gru    = (const float*)d_in[12];
  const float* b_gru_in = (const float*)d_in[13];
  const float* b_gru_rec= (const float*)d_in[14];
  const float* ln_lin_g = (const float*)d_in[15];
  const float* ln_lin_b = (const float*)d_in[16];
  const float* W1       = (const float*)d_in[17];
  const float* b1       = (const float*)d_in[18];
  const float* W2       = (const float*)d_in[19];
  const float* b2       = (const float*)d_in[20];

  char* ws = (char*)d_ws;
  size_t off = 0;
  auto alloc = [&](size_t bytes) -> void* {
    void* p = ws + off;
    off = (off + bytes + 255) & ~(size_t)255;
    return p;
  };
  __bf16* xkey   = (__bf16*)alloc((size_t)B_ * N_ * D_ * 2);   // 64 MB
  __bf16* xvalT  = (__bf16*)alloc((size_t)B_ * N_ * D_ * 2);   // 64 MB
  float*  slots  = (float*)alloc((size_t)B_ * K_ * D_ * 4);
  float*  raw    = (float*)alloc((size_t)B_ * K_ * D_ * 4);
  float*  sumval = (float*)alloc((size_t)B_ * D_ * 4);
  float*  csum   = (float*)alloc((size_t)B_ * K_ * 4);

  init_kernel<<<64, 256, 0, stream>>>(noise, loc, log_scl, slots, sumval, raw, csum);
  proj_kernel<<<4096, 256, 0, stream>>>(x, ln_in_g, ln_in_b, W_key, W_val, xkey, xvalT, sumval);
  for (int step = 0; step < 3; ++step) {
    attn_kernel<<<B_ * 16, 256, 0, stream>>>(slots, ln_q_g, ln_q_b, W_qry,
                                             xkey, xvalT, raw, csum);
    combine_kernel<<<B_ * K_, 64, 0, stream>>>(raw, csum, sumval, slots, W_gru, U_gru,
                                               b_gru_in, b_gru_rec, ln_lin_g, ln_lin_b,
                                               W1, b1, W2, b2);
  }
  hipMemcpyAsync(d_out, slots, (size_t)B_ * K_ * D_ * 4, hipMemcpyDeviceToDevice, stream);
}
